// AlignMixModel_1692217114928
// MI455X (gfx1250) — compile-verified, matrix operand order checked by reading, not run on hardware
//
#include <hip/hip_runtime.h>
#include <math.h>

typedef __attribute__((ext_vector_type(16))) _Float16 v16h;
typedef __attribute__((ext_vector_type(8)))  _Float16 v8h;
typedef __attribute__((ext_vector_type(8)))  float    v8f;

// ---------------- constants ----------------
#define Bsz   32
#define Cin   128
#define Cmid  256
#define Emb   128
#define S1    1024   // 32*32
#define S2    256    // 16*16
#define NCLS  100
#define LAM   0.7f
#define ILAM  0.3f

// padded spatial extents (zero halo, channel-last)
#define XP    34     // 32 + 1 halo each side
#define CP    18     // 16 + 1 halo each side

// f16 WMMA fragment (ISA 7.12.2, 16-bit 16x32): for a lane, the 16 elements are
// K = base+0..7 (VGPR0-3) and K = base+16..23 (VGPR4-7), base = (lane&16)?8:0.
// With K-contiguous data this is exactly two 16-byte loads.
__device__ __forceinline__ v16h load_frag16(const _Float16* __restrict__ p, int lane) {
  const int base = (lane & 16) ? 8 : 0;
  v8h lo = *(const v8h*)(p + base);
  v8h hi = *(const v8h*)(p + base + 16);
  v16h r;
#pragma unroll
  for (int e = 0; e < 8; ++e) { r[e] = lo[e]; r[e + 8] = hi[e]; }
  return r;
}

// =====================================================================
// prep: weights (O,I,3,3) f32 -> (O,ky,kx,I) f16 (IC contiguous)
// =====================================================================
__global__ void prep_wT(const float* __restrict__ w, _Float16* __restrict__ wt,
                        int O, int I) {
  int t = blockIdx.x * blockDim.x + threadIdx.x;
  int total = O * I * 9;
  if (t >= total) return;
  int i = t % I;
  int rest = t / I;
  int kidx = rest % 9;
  int o = rest / 9;
  wt[t] = (_Float16)w[((size_t)o * I + i) * 9 + kidx];
}

// prep: x (B,C,32,32) f32 -> xTp (B,34,34,C) f16 channel-last with zero halo
__global__ void prep_xT(const float* __restrict__ x, _Float16* __restrict__ xt) {
  int t = blockIdx.x * blockDim.x + threadIdx.x;
  if (t >= Bsz * S1 * Cin) return;
  int c = t & 127;
  int s = (t >> 7) & 1023;
  int b = t >> 17;
  int iy = s >> 5, ix = s & 31;
  xt[(((size_t)b * XP + iy + 1) * XP + ix + 1) * Cin + c] =
      (_Float16)x[((size_t)b * Cin + c) * S1 + s];
}

__global__ void zero_kernel(float* __restrict__ p, int n) {
  int t = blockIdx.x * blockDim.x + threadIdx.x;
  if (t < n) p[t] = 0.f;
}

// =====================================================================
// conv1: c = relu(conv(x, W_enc, stride2, SAME))
// xTp:(B,34,34,128)f16 -> cTp:(B,18,18,256)f16 (zero halo); 1 wave / 16x16 tile
// inner loop: 4x global_load_b128 + 1x v_wmma (no masking, halo is zero)
// =====================================================================
__global__ __launch_bounds__(32) void conv1_wmma(const _Float16* __restrict__ xTp,
                                                 const _Float16* __restrict__ WencT,
                                                 _Float16* __restrict__ cTp) {
  int lane = threadIdx.x;
  int oc0 = blockIdx.x * 16;          // 0..255
  int p0  = blockIdx.y * 16;          // 0..255
  int b   = blockIdx.z;
  int l15 = lane & 15;                // A row (oc) & B/D col (spatial)
  int n   = p0 + l15;
  int oy  = n >> 4, ox = n & 15;
  const _Float16* wbase = WencT + (size_t)(oc0 + l15) * (9 * Cin);
  const _Float16* xbase = xTp + (size_t)b * XP * XP * Cin;
  v8f acc = {};
  for (int ky = 0; ky < 3; ++ky) {
    for (int kx = 0; kx < 3; ++kx) {
      // padded coords: iy_pad = 2*oy+ky-1+1, ix_pad = 2*ox+kx-1+1
      const _Float16* bp = xbase + (size_t)((2 * oy + ky) * XP + 2 * ox + kx) * Cin;
      const _Float16* ap = wbase + (ky * 3 + kx) * Cin;
      __builtin_prefetch(ap, 0, 1);
#pragma unroll
      for (int kc = 0; kc < Cin; kc += 32) {
        v16h A  = load_frag16(ap + kc, lane);
        v16h Bv = load_frag16(bp + kc, lane);
        acc = __builtin_amdgcn_wmma_f32_16x16x32_f16(false, A, false, Bv,
                                                     (short)0, acc, false, false);
      }
    }
  }
  int hib = (lane & 16) ? 8 : 0;
  v8h o;
#pragma unroll
  for (int r = 0; r < 8; ++r) {
    float v = acc[r];
    o[r] = (_Float16)(v > 0.f ? v : 0.f);
  }
  *(v8h*)(cTp + (((size_t)b * CP + oy + 1) * CP + ox + 1) * Cmid + oc0 + hib) = o;
}

// =====================================================================
// conv2: f = relu(conv(c, W_feat, stride1, SAME)); only spatial mean needed
// cTp:(B,18,18,256)f16 -> atomic featsum (B,128) f32
// =====================================================================
__global__ __launch_bounds__(32) void conv2_wmma(const _Float16* __restrict__ cTp,
                                                 const _Float16* __restrict__ WfeatT,
                                                 float* __restrict__ featsum) {
  int lane = threadIdx.x;
  int oc0 = blockIdx.x * 16;          // 0..127
  int p0  = blockIdx.y * 16;          // 0..255
  int b   = blockIdx.z;
  int l15 = lane & 15;
  int n   = p0 + l15;
  int oy  = n >> 4, ox = n & 15;
  const _Float16* wbase = WfeatT + (size_t)(oc0 + l15) * (9 * Cmid);
  const _Float16* cbase = cTp + (size_t)b * CP * CP * Cmid;
  v8f acc = {};
  for (int ky = 0; ky < 3; ++ky) {
    for (int kx = 0; kx < 3; ++kx) {
      const _Float16* bp = cbase + (size_t)((oy + ky) * CP + ox + kx) * Cmid;
      const _Float16* ap = wbase + (ky * 3 + kx) * Cmid;
      __builtin_prefetch(ap, 0, 1);
#pragma unroll
      for (int kc = 0; kc < Cmid; kc += 32) {
        v16h A  = load_frag16(ap + kc, lane);
        v16h Bv = load_frag16(bp + kc, lane);
        acc = __builtin_amdgcn_wmma_f32_16x16x32_f16(false, A, false, Bv,
                                                     (short)0, acc, false, false);
      }
    }
  }
  int hib = (lane & 16) ? 8 : 0;
#pragma unroll
  for (int r = 0; r < 8; ++r) {
    float v = acc[r];
    v = v > 0.f ? v : 0.f;
    atomicAdd(&featsum[b * Emb + oc0 + r + hib], v);
  }
}

// =====================================================================
// deconv (conv_transpose stride2 SAME), decomposed into 4 output-parity
// classes: (0,0)->1 tap, (0,1)/(1,0)->2 taps, (1,1)->4 taps. Accumulates MSE.
//   out(2m+py, 2n+px) = sum_taps c[m+dy][n+dx] . W[ky][kx]
//   py==0: {ky=1,dy=0};  py==1: {ky=0,dy=0},{ky=2,dy=1}   (same for x)
// =====================================================================
__global__ __launch_bounds__(32) void deconv_wmma(const _Float16* __restrict__ cTp,
                                                  const _Float16* __restrict__ WdecT,
                                                  const float* __restrict__ xref,
                                                  float* __restrict__ errAcc) {
  int lane = threadIdx.x;
  int oc0 = blockIdx.x * 16;            // 0..127
  int ntile = blockIdx.y & 15;          // 16 tiles per class (16x16 class grid)
  int cls   = blockIdx.y >> 4;          // 0..3
  int b     = blockIdx.z;
  int py = cls >> 1, px = cls & 1;
  int l15 = lane & 15;
  int q  = ntile * 16 + l15;            // class-grid linear position
  int m  = q >> 4, n = q & 15;          // class-grid coords (0..15)
  int oy = 2 * m + py, ox = 2 * n + px; // output pixel
  const _Float16* wbase = WdecT + (size_t)(oc0 + l15) * (9 * Cmid);
  const _Float16* cbase = cTp + (size_t)b * CP * CP * Cmid;

  int kys[2], dys[2], kxs[2], dxs[2];
  int nty, ntx;
  if (py) { nty = 2; kys[0] = 0; dys[0] = 0; kys[1] = 2; dys[1] = 1; }
  else    { nty = 1; kys[0] = 1; dys[0] = 0; }
  if (px) { ntx = 2; kxs[0] = 0; dxs[0] = 0; kxs[1] = 2; dxs[1] = 1; }
  else    { ntx = 1; kxs[0] = 1; dxs[0] = 0; }

  v8f acc = {};
  for (int a = 0; a < nty; ++a) {
    for (int t = 0; t < ntx; ++t) {
      const _Float16* bp = cbase + (size_t)((m + 1 + dys[a]) * CP + n + 1 + dxs[t]) * Cmid;
      const _Float16* ap = wbase + (kys[a] * 3 + kxs[t]) * Cmid;
      __builtin_prefetch(ap, 0, 1);
#pragma unroll
      for (int kc = 0; kc < Cmid; kc += 32) {
        v16h A  = load_frag16(ap + kc, lane);
        v16h Bv = load_frag16(bp + kc, lane);
        acc = __builtin_amdgcn_wmma_f32_16x16x32_f16(false, A, false, Bv,
                                                     (short)0, acc, false, false);
      }
    }
  }
  int hib = (lane & 16) ? 8 : 0;
  float lerr = 0.f;
#pragma unroll
  for (int r = 0; r < 8; ++r) {
    int ch = oc0 + r + hib;
    float d = acc[r] - xref[(size_t)b * Cin * S1 + (size_t)ch * S1 + oy * 32 + ox];
    lerr += d * d;
  }
  atomicAdd(errAcc, lerr);
}

// =====================================================================
// invn[b,s] = 1/max(||x[b,:,s]||, 1e-12)
// =====================================================================
__global__ void colnorm_kernel(const float* __restrict__ x, float* __restrict__ invn) {
  int t = blockIdx.x * blockDim.x + threadIdx.x;
  if (t >= Bsz * S1) return;
  int b = t >> 10, s = t & 1023;
  const float* xp = x + (size_t)b * Cin * S1 + s;
  float ss = 0.f;
  for (int c = 0; c < Cin; ++c) { float v = xp[(size_t)c * S1]; ss += v * v; }
  invn[t] = 1.0f / fmaxf(sqrtf(ss), 1e-12f);
}

// sx[b,c] = sum_s x[b,c,s]*invn[b,s];  meanx[b,c] = mean_s x[b,c,s]
__global__ void rowsum_kernel(const float* __restrict__ x, const float* __restrict__ invn,
                              float* __restrict__ sx, float* __restrict__ meanx) {
  int t = blockIdx.x * blockDim.x + threadIdx.x;
  if (t >= Bsz * Cin) return;
  int b = t >> 7;
  const float* xp = x + (size_t)t * S1;
  const float* ip = invn + b * S1;
  float sn = 0.f, sm = 0.f;
  for (int s = 0; s < S1; ++s) { float v = xp[s]; sn += v * ip[s]; sm += v; }
  sx[t] = sn;
  meanx[t] = sm * (1.0f / 1024.0f);
}

// =====================================================================
// finalize: 4 feature sets + 6 metric-loss terms + output assembly
//   feat_ma = lam*mean(xa) + (1-lam)*sxb/1024  (sinkhorn T total mass == 1)
// =====================================================================
__global__ __launch_bounds__(256) void finalize_kernel(
    const float* __restrict__ featsum_a, const float* __restrict__ featsum_b,
    const float* __restrict__ meana, const float* __restrict__ meanb,
    const float* __restrict__ sxa, const float* __restrict__ sxb,
    const float* __restrict__ err, const float* __restrict__ proxies,
    const int* __restrict__ la, const int* __restrict__ lb,
    float* __restrict__ feats, float* __restrict__ out) {
  int tid = threadIdx.x;
  for (int i = tid; i < Bsz * Emb; i += 256) {
    feats[i]                 = featsum_a[i] * (1.0f / 256.0f);
    feats[Bsz * Emb + i]     = featsum_b[i] * (1.0f / 256.0f);
    feats[2 * Bsz * Emb + i] = LAM * meana[i] + ILAM * sxb[i] * (1.0f / 1024.0f);
    feats[3 * Bsz * Emb + i] = LAM * meanb[i] + ILAM * sxa[i] * (1.0f / 1024.0f);
  }
  __threadfence();
  __syncthreads();

  __shared__ float ps[NCLS];
  __shared__ float xsc[128];
  __shared__ float lossAcc[6];
  if (tid < NCLS) {
    const float* pp = proxies + tid * Emb;
    float s = 0.f;
    for (int c = 0; c < Emb; ++c) s += pp[c] * pp[c];
    ps[tid] = 3.0f / fmaxf(sqrtf(s), 1e-12f);
  }
  if (tid < 128) {
    const float* xp = feats + tid * Emb;
    float s = 0.f;
    for (int c = 0; c < Emb; ++c) s += xp[c] * xp[c];
    xsc[tid] = 3.0f / fmaxf(sqrtf(s), 1e-12f);
  }
  if (tid < 6) lossAcc[tid] = 0.f;
  __syncthreads();

  if (tid < 192) {
    int combo = tid >> 5;   // 0:(a,la) 1:(b,lb) 2:(ma,la) 3:(ma,lb) 4:(mb,lb) 5:(mb,la)
    int i = tid & 31;
    int set;
    const int* lbl;
    switch (combo) {
      case 0: set = 0; lbl = la; break;
      case 1: set = 1; lbl = lb; break;
      case 2: set = 2; lbl = la; break;
      case 3: set = 2; lbl = lb; break;
      case 4: set = 3; lbl = lb; break;
      default: set = 3; lbl = la; break;
    }
    const float* X = feats + (size_t)set * Bsz * Emb + (size_t)i * Emb;
    float xs = xsc[set * 32 + i];
    float Dv[NCLS];
    float mx = -1e30f;
    for (int j = 0; j < NCLS; ++j) {
      const float* pp = proxies + j * Emb;
      float d = 0.f;
      for (int c = 0; c < Emb; ++c) d += X[c] * pp[c];
      float nD = -(18.0f - 2.0f * xs * ps[j] * d);
      Dv[j] = nD;
      mx = fmaxf(mx, nD);
    }
    float se = 0.f;
    for (int j = 0; j < NCLS; ++j) se += __expf(Dv[j] - mx);
    float lse = mx + __logf(se);
    float logp = Dv[lbl[i]] - lse;
    atomicAdd(&lossAcc[combo], -logp);
  }
  __syncthreads();
  if (tid == 0) {
    float lca  = lossAcc[0] * (1.0f / 32.0f);
    float lcb  = lossAcc[1] * (1.0f / 32.0f);
    float lcma = LAM * lossAcc[2] * (1.0f / 32.0f) + ILAM * lossAcc[3] * (1.0f / 32.0f);
    float lcmb = LAM * lossAcc[4] * (1.0f / 32.0f) + ILAM * lossAcc[5] * (1.0f / 32.0f);
    float inv  = 1.0f / ((float)Bsz * Cin * S1);
    float lxa = err[0] * inv, lxb = err[1] * inv;
    out[0] = lxa + lxb + lca + lcb + lcma + lcmb;
    out[1] = lxa; out[2] = lxb; out[3] = lca; out[4] = lcb; out[5] = lcma; out[6] = lcmb;
  }
}

// =====================================================================
extern "C" void kernel_launch(void* const* d_in, const int* in_sizes, int n_in,
                              void* d_out, int out_size, void* d_ws, size_t ws_size,
                              hipStream_t stream) {
  const float* xa      = (const float*)d_in[0];
  const float* xb      = (const float*)d_in[1];
  const int*   la      = (const int*)d_in[2];
  const int*   lb      = (const int*)d_in[3];
  const float* proxies = (const float*)d_in[4];
  const float* Wenc    = (const float*)d_in[5];
  const float* Wfeat   = (const float*)d_in[6];
  const float* Wdec    = (const float*)d_in[7];
  float* out = (float*)d_out;
  (void)ws_size; (void)in_sizes; (void)n_in; (void)out_size;

  char* base = (char*)d_ws;
  size_t off = 0;
  auto alloc = [&](size_t bytes) -> char* {
    char* p = base + off;
    off += (bytes + 255) & ~(size_t)255;
    return p;
  };
  const size_t nxp = (size_t)Bsz * XP * XP * Cin;   // padded x halves
  const size_t ncp = (size_t)Bsz * CP * CP * Cmid;  // padded c halves
  _Float16* xaTp   = (_Float16*)alloc(nxp * 2);
  _Float16* xbTp   = (_Float16*)alloc(nxp * 2);
  _Float16* cTpa   = (_Float16*)alloc(ncp * 2);
  _Float16* cTpb   = (_Float16*)alloc(ncp * 2);
  _Float16* WencT  = (_Float16*)alloc((size_t)Cmid * 9 * Cin * 2);
  _Float16* WfeatT = (_Float16*)alloc((size_t)Emb * 9 * Cmid * 2);
  _Float16* WdecT  = (_Float16*)alloc((size_t)Cin * 9 * Cmid * 2);
  float* accblk    = (float*)alloc((size_t)(Bsz * Emb * 2 + 64) * 4);
  float* featsum_a = accblk;
  float* featsum_b = accblk + Bsz * Emb;
  float* errAcc    = accblk + 2 * Bsz * Emb;
  float* invn_a    = (float*)alloc((size_t)Bsz * S1 * 4);
  float* invn_b    = (float*)alloc((size_t)Bsz * S1 * 4);
  float* sxa       = (float*)alloc((size_t)Bsz * Emb * 4);
  float* sxb       = (float*)alloc((size_t)Bsz * Emb * 4);
  float* meana     = (float*)alloc((size_t)Bsz * Emb * 4);
  float* meanb     = (float*)alloc((size_t)Bsz * Emb * 4);
  float* feats     = (float*)alloc((size_t)4 * Bsz * Emb * 4);

  // zero: atomic accumulators + the padded f16 buffers (halos must be 0)
  int nzero = Bsz * Emb * 2 + 64;
  zero_kernel<<<(nzero + 255) / 256, 256, 0, stream>>>(accblk, nzero);
  int nza = (int)(nxp / 2);   // halves -> f32 words (counts are even)
  zero_kernel<<<(nza + 255) / 256, 256, 0, stream>>>((float*)xaTp, nza);
  zero_kernel<<<(nza + 255) / 256, 256, 0, stream>>>((float*)xbTp, nza);
  int nzc = (int)(ncp / 2);
  zero_kernel<<<(nzc + 255) / 256, 256, 0, stream>>>((float*)cTpa, nzc);
  zero_kernel<<<(nzc + 255) / 256, 256, 0, stream>>>((float*)cTpb, nzc);

  // layout prep: padded channel-last f16 activations + IC-contiguous f16 weights
  int nx = Bsz * S1 * Cin;
  prep_xT<<<(nx + 255) / 256, 256, 0, stream>>>(xa, xaTp);
  prep_xT<<<(nx + 255) / 256, 256, 0, stream>>>(xb, xbTp);
  prep_wT<<<(Cmid * Cin * 9 + 255) / 256, 256, 0, stream>>>(Wenc, WencT, Cmid, Cin);
  prep_wT<<<(Emb * Cmid * 9 + 255) / 256, 256, 0, stream>>>(Wfeat, WfeatT, Emb, Cmid);
  prep_wT<<<(Cin * Cmid * 9 + 255) / 256, 256, 0, stream>>>(Wdec, WdecT, Cin, Cmid);

  // channel-norm reductions for the algebraically-reduced mixing path
  colnorm_kernel<<<(Bsz * S1 + 255) / 256, 256, 0, stream>>>(xa, invn_a);
  colnorm_kernel<<<(Bsz * S1 + 255) / 256, 256, 0, stream>>>(xb, invn_b);
  rowsum_kernel<<<(Bsz * Cin + 255) / 256, 256, 0, stream>>>(xa, invn_a, sxa, meana);
  rowsum_kernel<<<(Bsz * Cin + 255) / 256, 256, 0, stream>>>(xb, invn_b, sxb, meanb);

  // encoder conv1 (WMMA implicit GEMM, pure b128 loads, zero-halo)
  conv1_wmma<<<dim3(Cmid / 16, S2 / 16, Bsz), 32, 0, stream>>>(xaTp, WencT, cTpa);
  conv1_wmma<<<dim3(Cmid / 16, S2 / 16, Bsz), 32, 0, stream>>>(xbTp, WencT, cTpb);

  // encoder conv2 -> feature sums
  conv2_wmma<<<dim3(Emb / 16, S2 / 16, Bsz), 32, 0, stream>>>(cTpa, WfeatT, featsum_a);
  conv2_wmma<<<dim3(Emb / 16, S2 / 16, Bsz), 32, 0, stream>>>(cTpb, WfeatT, featsum_b);

  // decoder conv_transpose (parity-decomposed) -> reconstruction MSE
  deconv_wmma<<<dim3(Cin / 16, 64, Bsz), 32, 0, stream>>>(cTpa, WdecT, xa, errAcc);
  deconv_wmma<<<dim3(Cin / 16, 64, Bsz), 32, 0, stream>>>(cTpb, WdecT, xb, errAcc + 1);

  // metric losses + final assembly
  finalize_kernel<<<1, 256, 0, stream>>>(featsum_a, featsum_b, meana, meanb,
                                         sxa, sxb, errAcc, proxies, la, lb,
                                         feats, out);
}
